// GRUModel_20409684590892
// MI455X (gfx1250) — matrix operand order repaired and checked by reading, not compiled, not run on hardware
//
#include <hip/hip_runtime.h>
#include <hip/hip_bf16.h>

// GRU LM for MI455X (gfx1250): bf16 WMMA GEMMs with f32 accumulation,
// TDM (tensor_load_to_lds) double-buffered LDS staging.
// B=64 S=512 V=32000 E=512 H=1024 XD=512 C=1024
#define BD 64
#define SD 512
#define ED 512
#define HD 1024
#define XDD 512
#define CD 1024

typedef __attribute__((ext_vector_type(16))) __bf16 v16bf;
typedef __attribute__((ext_vector_type(8)))  float  v8f;
typedef __attribute__((ext_vector_type(4)))  unsigned v4u;
typedef __attribute__((ext_vector_type(8)))  unsigned v8u;

__device__ __forceinline__ unsigned short f2bf(float f) {
  unsigned u = __float_as_uint(f);
  u += 0x7FFFu + ((u >> 16) & 1u);   // round-to-nearest-even
  return (unsigned short)(u >> 16);
}

// Issue one 2D-tile TDM DMA: global[rows x 32 bf16, row stride strideElems]
// -> LDS at byte offset lds_off (rows packed contiguously, 64B/row).
// D# per CDNA5 ISA ch.8; 2-operand form (groups 2/3 NULL => 2D tensor).
__device__ __forceinline__ void tdm_load_2d(unsigned lds_off, const void* gptr,
                                            unsigned rows, unsigned strideElems) {
  unsigned long long ga = (unsigned long long)(uintptr_t)gptr;
  v4u g0;
  g0[0] = 1u;                                            // count=1 (valid user D#)
  g0[1] = lds_off;                                       // lds_addr (bytes)
  g0[2] = (unsigned)ga;                                  // global_addr[31:0]
  g0[3] = (unsigned)((ga >> 32) & 0x01FFFFFFu) | (2u << 30); // addr[56:32], type=2
  v8u g1;
  g1[0] = 1u << 16;          // workgroup_mask=0, data_size=1 (2 bytes)
  g1[1] = 32u << 16;         // tensor_dim0[15:0] = 32 (bits 79:48)
  g1[2] = rows << 16;        // tensor_dim1[15:0] = rows (bits 111:80)
  g1[3] = 32u << 16;         // tile_dim0 = 32 (bits 127:112)
  g1[4] = rows;              // tile_dim1 = rows, tile_dim2 = 0
  g1[5] = strideElems;       // tensor_dim0_stride[31:0] (data_size units)
  g1[6] = 0u;                // stride hi, dim1_stride lo
  g1[7] = 0u;
  asm volatile("tensor_load_to_lds %0, %1" :: "s"(g0), "s"(g1) : "memory");
}

struct GemmP {
  const unsigned short* A;    // [M,K] bf16 row-major
  const unsigned short* Bw;   // [N,K] bf16 row-major (we compute A·Bwᵀ)
  const float* bias;          // [N] or null
  const float* pre;           // pre-add matrix (x-projection), row stride ldpre
  long  ldpre;
  float* out_f32;             // mode 0: C ; mode 2: H_all slot t
  unsigned short* out_bf16;   // mode 1: rh ; mode 2: Hall_bf16 slot t ; mode 4: Y
  const float* h_prev;        // mode 1/2: previous hidden state (f32)
  float* z_buf;               // mode 1 write / mode 2 read
  float* out_scatter;         // mode 3: final [B,S,C] output
  int M, N, K, mode;
};

#define LA_ELEMS (64 * 32)
#define LB_ELEMS (128 * 32)
#define BUF_ELEMS (LA_ELEMS + LB_ELEMS)

__global__ __launch_bounds__(256) void wmma_gemm(GemmP p) {
  __shared__ unsigned short lds[2 * BUF_ELEMS];   // 2 x 12KB, double-buffered
  const int tid  = threadIdx.x;
  const int wave = tid >> 5, lane = tid & 31;
  const int l15  = lane & 15, lhi = lane >> 4;
  const int waveM = (wave & 1) * 32, waveN = (wave >> 1) * 32;
  const long blockM = (long)blockIdx.y * 64;
  const long blockN = (long)blockIdx.x * 128;
  const int  K = p.K;

  // LDS byte offsets of the two staging buffers (flat->LDS = low 32 bits)
  unsigned abase[2], bbase[2];
  abase[0] = (unsigned)(uintptr_t)&lds[0];
  bbase[0] = (unsigned)(uintptr_t)&lds[LA_ELEMS];
  abase[1] = (unsigned)(uintptr_t)&lds[BUF_ELEMS];
  bbase[1] = (unsigned)(uintptr_t)&lds[BUF_ELEMS + LA_ELEMS];

  const unsigned short* Ab = p.A  + blockM * (long)K;   // block's A rows
  const unsigned short* Bb = p.Bw + blockN * (long)K;   // block's B rows

  if (p.pre)  // warm L2 for the epilogue pre-add rows (global_prefetch_b8)
    __builtin_prefetch(&p.pre[(blockM + (tid & 63)) * p.ldpre + blockN], 0, 0);

  v8f acc[2][2] = {};
  union Frag { v16bf v; unsigned u[8]; };

  // Prologue: DMA tiles for k0=0 into buffer 0
  if (wave == 0) {
    tdm_load_2d(abase[0], Ab, 64, (unsigned)K);
    tdm_load_2d(bbase[0], Bb, 128, (unsigned)K);
    __builtin_amdgcn_s_wait_tensorcnt(0);
  }
  __syncthreads();

  int cur = 0;
  for (int k0 = 0; k0 < K; k0 += 32) {
    // Overlap: DMA next K-slab into the other buffer while we compute
    if (wave == 0 && k0 + 32 < K) {
      tdm_load_2d(abase[cur ^ 1], Ab + (k0 + 32), 64, (unsigned)K);
      tdm_load_2d(bbase[cur ^ 1], Bb + (k0 + 32), 128, (unsigned)K);
    }

    const unsigned short* lA = &lds[cur * BUF_ELEMS];
    const unsigned short* lB = lA + LA_ELEMS;

    // Fragments: contiguous k order (k = half*16 + s), same for A and B.
    Frag a[2], b[2];
    #pragma unroll
    for (int s = 0; s < 2; ++s) {
      int m = waveM + s * 16 + l15;
      #pragma unroll
      for (int v = 0; v < 8; ++v) {        // A: K pair = half*16 + 2v
        int kk = lhi * 16 + v * 2;
        a[s].u[v] = *(const unsigned*)&lA[m * 32 + kk];
      }
      int n = waveN + s * 16 + l15;
      #pragma unroll
      for (int v = 0; v < 8; ++v) {        // B: K pair = half*16 + 2v
        int kk = lhi * 16 + v * 2;
        b[s].u[v] = *(const unsigned*)&lB[n * 32 + kk];
      }
    }
    #pragma unroll
    for (int i = 0; i < 2; ++i)
      #pragma unroll
      for (int j = 0; j < 2; ++j)
        acc[i][j] = __builtin_amdgcn_wmma_f32_16x16x32_bf16(
            false, a[i].v, false, b[j].v, (short)0, acc[i][j], false, false);

    __syncthreads();                                   // everyone done reading cur
    if (wave == 0 && k0 + 32 < K)
      __builtin_amdgcn_s_wait_tensorcnt(0);            // next buffer landed
    __syncthreads();
    cur ^= 1;
  }

  // Epilogue: C layout VGPR v -> row (lane>>4)*8 + v, col lane&15
  #pragma unroll
  for (int i = 0; i < 2; ++i) {
    #pragma unroll
    for (int j = 0; j < 2; ++j) {
      #pragma unroll
      for (int v = 0; v < 8; ++v) {
        long row = blockM + waveM + i * 16 + lhi * 8 + v;
        long col = blockN + waveN + j * 16 + l15;
        float val = acc[i][j][v];
        if (p.bias) val += p.bias[col];
        if (p.pre)  val += p.pre[row * p.ldpre + col];
        switch (p.mode) {
          case 0:  // plain f32 store (Gx precompute)
            p.out_f32[row * (long)p.N + col] = val;
            break;
          case 1: {  // z,r gates: sigmoid; z->z_buf, r -> rh = bf16(r*h)
            float s = 1.0f / (1.0f + __expf(-val));
            if (col < HD) {
              p.z_buf[row * HD + col] = s;
            } else {
              long c2 = col - HD;
              p.out_bf16[row * HD + c2] = f2bf(s * p.h_prev[row * HD + c2]);
            }
            break;
          }
          case 2: {  // candidate + state update
            float hc = tanhf(val);
            float z  = p.z_buf[row * HD + col];
            float h  = p.h_prev[row * HD + col];
            float hn = (1.0f - z) * h + z * hc;
            p.out_f32[row * HD + col]  = hn;        // H_all[t]
            p.out_bf16[row * HD + col] = f2bf(hn);  // bf16 state (next-step A / head A)
            break;
          }
          case 3: {  // final logits: rows are t*B+b, scatter to [B,S,C]
            long t = row / BD, b = row % BD;
            p.out_scatter[(b * (long)SD + t) * (long)CD + col] = val;
            break;
          }
          case 4:  // bf16 store (Y head intermediate)
            p.out_bf16[row * (long)p.N + col] = f2bf(val);
            break;
        }
      }
    }
  }
}

__global__ void k_embed(const int* X, const float* embed, unsigned short* out, long ntok) {
  long idx = (long)blockIdx.x * blockDim.x + threadIdx.x;
  if (idx >= ntok * (long)ED) return;
  long tok = idx / ED, e = idx % ED;
  out[idx] = f2bf(embed[(long)X[tok] * ED + e]);
}

__global__ void k_f2bf(const float* s, unsigned short* d, long n) {
  long i = (long)blockIdx.x * blockDim.x + threadIdx.x;
  if (i < n) d[i] = f2bf(s[i]);
}

__global__ void k_copyf(const float* s, float* d, long n) {
  long i = (long)blockIdx.x * blockDim.x + threadIdx.x;
  if (i < n) d[i] = s[i];
}

extern "C" void kernel_launch(void* const* d_in, const int* in_sizes, int n_in,
                              void* d_out, int out_size, void* d_ws, size_t ws_size,
                              hipStream_t stream) {
  const int*   X     = (const int*)d_in[0];
  const float* h0    = (const float*)d_in[1];
  const float* embed = (const float*)d_in[2];
  const float* Wzh = (const float*)d_in[3];  const float* bzh = (const float*)d_in[4];
  const float* Wzx = (const float*)d_in[5];  const float* bzx = (const float*)d_in[6];
  const float* Wrh = (const float*)d_in[7];  const float* brh = (const float*)d_in[8];
  const float* Wrx = (const float*)d_in[9];  const float* brx = (const float*)d_in[10];
  const float* Wuh = (const float*)d_in[11]; const float* buh = (const float*)d_in[12];
  const float* Wux = (const float*)d_in[13]; const float* bux = (const float*)d_in[14];
  const float* Wxh = (const float*)d_in[15]; const float* bxh = (const float*)d_in[16];
  const float* Wc  = (const float*)d_in[17]; const float* bc  = (const float*)d_in[18];
  float* out = (float*)d_out;

  const long NT = (long)BD * SD;  // 32768 tokens
  char* ws = (char*)d_ws;
  auto carve = [&](size_t bytes) -> char* {
    char* r = ws;
    ws += (bytes + 255) & ~(size_t)255;
    return r;
  };
  float*          Gx      = (float*)carve(NT * 3072 * 4);            // x-projections z|r|u
  float*          Hall    = (float*)carve((long)SD * BD * HD * 4);   // f32 states
  unsigned short* HallBf  = (unsigned short*)carve((long)SD * BD * HD * 2);
  unsigned short* XE      = (unsigned short*)carve(NT * ED * 2);     // bf16 embeddings
  unsigned short* Ybf     = (unsigned short*)carve(NT * XDD * 2);
  float*          zbuf    = (float*)carve((long)BD * HD * 4);
  unsigned short* rh      = (unsigned short*)carve((long)BD * HD * 2);
  unsigned short* h0bf    = (unsigned short*)carve((long)BD * HD * 2);
  unsigned short* WxC     = (unsigned short*)carve(3072L * ED * 2);  // [Wzx;Wrx;Wux]
  unsigned short* Wzr     = (unsigned short*)carve(2048L * HD * 2);  // [Wzh;Wrh]
  unsigned short* WuhB    = (unsigned short*)carve((long)HD * HD * 2);
  unsigned short* WxhB    = (unsigned short*)carve((long)XDD * HD * 2);
  unsigned short* WcB     = (unsigned short*)carve((long)CD * XDD * 2);
  float*          bxC     = (float*)carve(3072 * 4);                 // [bzx;brx;bux]
  float*          bzrC    = (float*)carve(2048 * 4);                 // [bzh;brh]

  auto cvt = [&](const float* s, unsigned short* d, long n) {
    k_f2bf<<<dim3((unsigned)((n + 255) / 256)), 256, 0, stream>>>(s, d, n);
  };
  auto cpy = [&](const float* s, float* d, long n) {
    k_copyf<<<dim3((unsigned)((n + 255) / 256)), 256, 0, stream>>>(s, d, n);
  };

  // Weight / bias prep (bf16 conversion + concatenation)
  cvt(Wzx, WxC,               1024L * 512);
  cvt(Wrx, WxC + 1024L * 512, 1024L * 512);
  cvt(Wux, WxC + 2048L * 512, 1024L * 512);
  cvt(Wzh, Wzr,                1024L * 1024);
  cvt(Wrh, Wzr + 1024L * 1024, 1024L * 1024);
  cvt(Wuh, WuhB, (long)HD * HD);
  cvt(Wxh, WxhB, (long)XDD * HD);
  cvt(Wc,  WcB,  (long)CD * XDD);
  cvt(h0,  h0bf, (long)BD * HD);
  cpy(bzx, bxC,        1024);
  cpy(brx, bxC + 1024, 1024);
  cpy(bux, bxC + 2048, 1024);
  cpy(bzh, bzrC,        1024);
  cpy(brh, bzrC + 1024, 1024);

  // Embedding gather -> bf16  [NT, E]
  k_embed<<<dim3((unsigned)((NT * ED + 255) / 256)), 256, 0, stream>>>(X, embed, XE, NT);

  auto gemm = [&](const unsigned short* A, const unsigned short* Bw,
                  const float* bias, const float* pre, long ldpre,
                  float* of, unsigned short* ob, const float* hp,
                  float* zb, float* osc, int M, int N, int K, int mode) {
    GemmP p{A, Bw, bias, pre, ldpre, of, ob, hp, zb, osc, M, N, K, mode};
    dim3 g((unsigned)(N / 128), (unsigned)(M / 64));
    wmma_gemm<<<g, 256, 0, stream>>>(p);
  };

  // Phase A: time-parallel x-projections. Gx[NT,3072] = XE·WxCᵀ + bxC
  gemm(XE, WxC, bxC, nullptr, 0, Gx, nullptr, nullptr, nullptr, nullptr,
       (int)NT, 3072, ED, 0);

  // Phase B: sequential recurrence, 2 fused GEMM launches per step
  for (int t = 0; t < SD; ++t) {
    const unsigned short* hA = t ? HallBf + (long)(t - 1) * BD * HD : h0bf;
    const float*          hp = t ? Hall   + (long)(t - 1) * BD * HD : h0;
    // gates: sigmoid(h·[Wzh;Wrh]ᵀ + [bzh;brh] + Gx[:, :2048]); z->zbuf, rh=bf16(r*h)
    gemm(hA, Wzr, bzrC, Gx + (long)t * 3072, (long)SD * 3072,
         nullptr, rh, hp, zbuf, nullptr, BD, 2048, HD, 1);
    // candidate + update: h' = (1-z)h + z·tanh(rh·Wuhᵀ + buh + Gx[:,2048:])
    gemm(rh, WuhB, buh, Gx + (long)t * 3072 + 2048, (long)SD * 3072,
         Hall + (long)t * BD * HD, HallBf + (long)t * BD * HD, hp, zbuf, nullptr,
         BD, HD, HD, 2);
  }

  // Phase C: batched output head over all (t,b) rows
  gemm(HallBf, WxhB, bxh, nullptr, 0, nullptr, Ybf, nullptr, nullptr, nullptr,
       (int)NT, XDD, HD, 4);
  gemm(Ybf, WcB, bc, nullptr, 0, nullptr, nullptr, nullptr, nullptr, out,
       (int)NT, CD, XDD, 3);
}
